// RWKVBlock_23605140259526
// MI455X (gfx1250) — matrix operand order repaired and checked by reading, not compiled, not run on hardware
//
#include <hip/hip_runtime.h>
#include <hip/hip_bf16.h>

typedef __attribute__((ext_vector_type(16))) __bf16 bf16x16;
typedef __attribute__((ext_vector_type(8)))  __bf16 bf16x8;
typedef __attribute__((ext_vector_type(8)))  float  f32x8;
typedef __attribute__((ext_vector_type(4)))  int    i32x4;

#define ASG __attribute__((address_space(1)))
#define ASL __attribute__((address_space(3)))

#define LN_EPS  1e-5f
#define WKV_EPS 1e-8f

// problem constants
#define BB 4
#define SS 2048
#define DD 1024
#define FF 4096
#define MROWS (BB*SS)          // 8192

// WKV chunked-scan config
#define SC 32                  // chunks along S
#define SL (SS/SC)             // 64 steps per chunk

// GEMM tiling
#define BM 128
#define BN 128
#define BK 32
#define LDT 40                 // LDS row stride (bf16), 32 + 8 pad, keeps 16B align

enum { EP_STORE=0, EP_SIGMOID=1, EP_EWK=2, EP_ADDX=3, EP_RELU2SPLIT=4, EP_FINAL=5 };

__device__ __forceinline__ void split2(float x, __bf16& h, __bf16& l) {
    h = (__bf16)x;
    l = (__bf16)(x - (float)h);
}

__device__ __forceinline__ bf16x16 ld16(const __bf16* p0, const __bf16* p1) {
    bf16x8 a = *(const bf16x8*)p0;
    bf16x8 b = *(const bf16x8*)p1;
    bf16x16 r;
#pragma unroll
    for (int i = 0; i < 8; ++i) { r[i] = a[i]; r[8+i] = b[i]; }
    return r;
}

__device__ __forceinline__ f32x8 wmma_bf16(bf16x16 a, bf16x16 b, f32x8 c) {
    return __builtin_amdgcn_wmma_f32_16x16x32_bf16(false, a, false, b, (short)0, c, false, false);
}

// --- CDNA5 async global->LDS copy (16B), with sync fallback --------------
__device__ __forceinline__ void async_cp16(__bf16* lds, const __bf16* g) {
#if __has_builtin(__builtin_amdgcn_global_load_async_to_lds_b128)
    __builtin_amdgcn_global_load_async_to_lds_b128(
        (ASG i32x4*)(unsigned long long)(const void*)g,
        (ASL i32x4*)lds, 0, 0);
#else
    *(bf16x8*)lds = *(const bf16x8*)g;
#endif
}

__device__ __forceinline__ void wait_async0() {
#if __has_builtin(__builtin_amdgcn_s_wait_asynccnt)
    __builtin_amdgcn_s_wait_asynccnt(0);
#elif __has_builtin(__builtin_amdgcn_global_load_async_to_lds_b128)
    asm volatile("s_wait_asynccnt 0" ::: "memory");
#endif
}

// ---------------------------------------------------------------------------
// fp32 -> (bf16 hi, bf16 lo) elementwise split
// ---------------------------------------------------------------------------
__global__ __launch_bounds__(256)
void k_split(const float* __restrict__ src, __bf16* __restrict__ h,
             __bf16* __restrict__ l, int n) {
    int i = blockIdx.x * blockDim.x + threadIdx.x;
    if (i < n) {
        __bf16 hh, ll;
        split2(src[i], hh, ll);
        h[i] = hh; l[i] = ll;
    }
}

// ---------------------------------------------------------------------------
// LayerNorm over D + bf16 hi/lo split. One block (256 thr) per row.
// ---------------------------------------------------------------------------
__global__ __launch_bounds__(256)
void k_ln_split(const float* __restrict__ x, const float* __restrict__ g,
                const float* __restrict__ b, __bf16* __restrict__ xh,
                __bf16* __restrict__ xl) {
    __shared__ float red[32];
    const int row = blockIdx.x;
    const float* xr = x + (size_t)row * DD;
    float s = 0.f, s2 = 0.f;
    for (int i = threadIdx.x; i < DD; i += 256) {
        float v = xr[i]; s += v; s2 += v * v;
    }
#pragma unroll
    for (int off = 16; off; off >>= 1) {
        s  += __shfl_down(s,  off, 32);
        s2 += __shfl_down(s2, off, 32);
    }
    const int wid = threadIdx.x >> 5, lid = threadIdx.x & 31;
    if (lid == 0) { red[wid] = s; red[8 + wid] = s2; }
    __syncthreads();
    if (threadIdx.x == 0) {
        float ts = 0.f, ts2 = 0.f;
        for (int w = 0; w < 8; ++w) { ts += red[w]; ts2 += red[8 + w]; }
        float mu = ts / (float)DD;
        float var = ts2 / (float)DD - mu * mu;
        red[16] = mu;
        red[17] = rsqrtf(var + LN_EPS);
    }
    __syncthreads();
    const float mu = red[16], rs = red[17];
    for (int i = threadIdx.x; i < DD; i += 256) {
        float v = (xr[i] - mu) * rs * g[i] + b[i];
        size_t o = (size_t)row * DD + i;
        __bf16 hh, ll; split2(v, hh, ll);
        xh[o] = hh; xl[o] = ll;
    }
}

// ---------------------------------------------------------------------------
// Stage one 128x32 hi/lo tile pair of A and W into LDS via async copies.
// ---------------------------------------------------------------------------
__device__ __forceinline__
void stage_tiles(const __bf16* __restrict__ Ah, const __bf16* __restrict__ Al,
                 const __bf16* __restrict__ Bh, const __bf16* __restrict__ Bl,
                 __bf16* sAh, __bf16* sAl, __bf16* sBh, __bf16* sBl,
                 int tid, int m0, int n0, int kc, int K) {
#pragma unroll
    for (int p = 0; p < 2; ++p) {
        const int e = (p * 256 + tid) * 8;     // element in 128x32 tile
        const int r = e >> 5, c = e & 31;
        const size_t ga = (size_t)(m0 + r) * K + kc + c;
        const size_t gb = (size_t)(n0 + r) * K + kc + c;
        async_cp16(&sAh[r * LDT + c], &Ah[ga]);
        async_cp16(&sAl[r * LDT + c], &Al[ga]);
        async_cp16(&sBh[r * LDT + c], &Bh[gb]);
        async_cp16(&sBl[r * LDT + c], &Bl[gb]);
    }
}

// ---------------------------------------------------------------------------
// bf16x3 WMMA GEMM: C[M,N] = A[M,K] * W[N,K]^T  (+ fused epilogue)
// 256 thr = 8 waves (4x2), 128x128 C tile, double-buffered LDS, async copies.
// ---------------------------------------------------------------------------
template <int EP>
__global__ __launch_bounds__(256)
void k_gemm(const __bf16* __restrict__ Ah, const __bf16* __restrict__ Al,
            const __bf16* __restrict__ Bh, const __bf16* __restrict__ Bl,
            int N, int K,
            float* __restrict__ Cout,
            __bf16* __restrict__ CoutH, __bf16* __restrict__ CoutL,
            const float* __restrict__ aux0, const float* __restrict__ aux1) {
    __shared__ __bf16 sAh[2][BM * LDT], sAl[2][BM * LDT];
    __shared__ __bf16 sBh[2][BN * LDT], sBl[2][BN * LDT];

    const int tid  = threadIdx.x;
    const int wave = tid >> 5, lane = tid & 31;
    const int wm = wave >> 1, wn = wave & 1;       // 4 x 2 wave grid
    const int lrow = lane & 15, lhi = lane >> 4;
    const int m0 = blockIdx.y * BM, n0 = blockIdx.x * BN;

    f32x8 acc[2][4];
#pragma unroll
    for (int mi = 0; mi < 2; ++mi)
#pragma unroll
        for (int ni = 0; ni < 4; ++ni)
#pragma unroll
            for (int e = 0; e < 8; ++e) acc[mi][ni][e] = 0.f;

    stage_tiles(Ah, Al, Bh, Bl, sAh[0], sAl[0], sBh[0], sBl[0],
                tid, m0, n0, 0, K);
    int buf = 0;
    for (int kc = 0; kc < K; kc += BK) {
        wait_async0();         // our async writes into `buf` done
        __syncthreads();       // everyone's done (also fences fallback path)
        if (kc + BK < K)
            stage_tiles(Ah, Al, Bh, Bl, sAh[buf ^ 1], sAl[buf ^ 1],
                        sBh[buf ^ 1], sBl[buf ^ 1], tid, m0, n0, kc + BK, K);

        // A frag: lane<16 holds K {0..7,16..23}, lane>=16 holds {8..15,24..31}
        bf16x16 afh[2], afl[2], bfh[4], bfl[4];
#pragma unroll
        for (int mi = 0; mi < 2; ++mi) {
            const int r = (wm * 32 + mi * 16 + lrow) * LDT;
            afh[mi] = ld16(&sAh[buf][r + lhi * 8], &sAh[buf][r + 16 + lhi * 8]);
            afl[mi] = ld16(&sAl[buf][r + lhi * 8], &sAl[buf][r + 16 + lhi * 8]);
        }
        // B frag: lane%16 = output col, lane<16 holds K {0..7,16..23}, lane>=16 holds {8..15,24..31}
#pragma unroll
        for (int ni = 0; ni < 4; ++ni) {
            const int r = (wn * 64 + ni * 16 + lrow) * LDT;
            bfh[ni] = ld16(&sBh[buf][r + lhi * 8], &sBh[buf][r + 16 + lhi * 8]);
            bfl[ni] = ld16(&sBl[buf][r + lhi * 8], &sBl[buf][r + 16 + lhi * 8]);
        }
#pragma unroll
        for (int mi = 0; mi < 2; ++mi)
#pragma unroll
            for (int ni = 0; ni < 4; ++ni) {
                acc[mi][ni] = wmma_bf16(afh[mi], bfh[ni], acc[mi][ni]);
                acc[mi][ni] = wmma_bf16(afh[mi], bfl[ni], acc[mi][ni]);
                acc[mi][ni] = wmma_bf16(afl[mi], bfh[ni], acc[mi][ni]);
            }
        buf ^= 1;
    }

    // epilogue: C/D layout -> row = m0 + wm*32 + mi*16 + j + 8*lhi, col = lrow
#pragma unroll
    for (int mi = 0; mi < 2; ++mi)
#pragma unroll
        for (int ni = 0; ni < 4; ++ni)
#pragma unroll
            for (int j = 0; j < 8; ++j) {
                const int r = m0 + wm * 32 + mi * 16 + j + lhi * 8;
                const int c = n0 + wn * 64 + ni * 16 + lrow;
                const size_t o = (size_t)r * N + c;
                const float v = acc[mi][ni][j];
                if (EP == EP_STORE) {
                    Cout[o] = v;
                } else if (EP == EP_SIGMOID) {
                    Cout[o] = 1.f / (1.f + __expf(-v));
                } else if (EP == EP_EWK) {
                    Cout[o] = __expf(aux0[o] - __expf(v));   // exp(k - exp(w))
                } else if (EP == EP_ADDX) {
                    Cout[o] = aux0[o] + v;
                } else if (EP == EP_RELU2SPLIT) {
                    float t = v > 0.f ? v : 0.f; t *= t;
                    __bf16 hh, ll; split2(t, hh, ll);
                    CoutH[o] = hh; CoutL[o] = ll;
                } else { // EP_FINAL
                    Cout[o] = aux0[o] + aux1[o] * v;
                }
            }
}

// ---------------------------------------------------------------------------
// WKV chunked scan. Linear recurrence s_t = a*s_{t-1} + u_t  =>
// state_after_chunk = a^SL * state_before + local(chunk with zero init).
// Phase 1: per-chunk locals (BB*SC*DD threads).
// Phase 2: 32-step prefix per channel; emits per-chunk init + final state.
// Phase 3: re-scan chunks with correct init; fuses a=sigmoid(r)*wkv + split.
// ---------------------------------------------------------------------------
__global__ __launch_bounds__(256)
void k_scan_phase1(const float* __restrict__ ewk, const float* __restrict__ v,
                   const float* __restrict__ td,
                   float* __restrict__ locN, float* __restrict__ locD) {
    const int gid = blockIdx.x * blockDim.x + threadIdx.x;
    if (gid >= BB * SC * DD) return;
    const int d = gid % DD;
    const int c = (gid / DD) % SC;
    const int b = gid / (DD * SC);
    const float decay = __expf(td[d]);
    float num = 0.f, den = 0.f;
    const size_t base = ((size_t)b * SS + (size_t)c * SL) * DD + d;
    for (int t = 0; t < SL; ++t) {
        const size_t o = base + (size_t)t * DD;
        const float e = ewk[o];
        num = decay * num + e * v[o];
        den = decay * den + e;
    }
    locN[gid] = num; locD[gid] = den;
}

__global__ __launch_bounds__(256)
void k_scan_phase2(const float* __restrict__ locN, const float* __restrict__ locD,
                   const float* __restrict__ td,
                   float* __restrict__ initN, float* __restrict__ initD,
                   float* __restrict__ state) {
    const int gid = blockIdx.x * blockDim.x + threadIdx.x;
    if (gid >= BB * DD) return;
    const int d = gid % DD, b = gid / DD;
    const float dl = __expf(td[d] * (float)SL);    // decay^SL
    float num = 0.f, den = 0.f;
#pragma unroll 4
    for (int c = 0; c < SC; ++c) {
        const size_t idx = ((size_t)b * SC + c) * DD + d;
        initN[idx] = num; initD[idx] = den;        // state entering chunk c
        num = dl * num + locN[idx];
        den = dl * den + locD[idx];
    }
    state[(size_t)b * 2 * DD + d]      = num;
    state[(size_t)b * 2 * DD + DD + d] = den;
}

__global__ __launch_bounds__(256)
void k_scan_phase3(const float* __restrict__ ewk, const float* __restrict__ v,
                   const float* __restrict__ sr, const float* __restrict__ td,
                   const float* __restrict__ initN, const float* __restrict__ initD,
                   __bf16* __restrict__ ah, __bf16* __restrict__ al) {
    const int gid = blockIdx.x * blockDim.x + threadIdx.x;
    if (gid >= BB * SC * DD) return;
    const int d = gid % DD;
    const int c = (gid / DD) % SC;
    const int b = gid / (DD * SC);
    const float decay = __expf(td[d]);
    float num = initN[gid], den = initD[gid];
    const size_t base = ((size_t)b * SS + (size_t)c * SL) * DD + d;
    for (int t = 0; t < SL; ++t) {
        const size_t o = base + (size_t)t * DD;
        const float e = ewk[o];
        num = decay * num + e * v[o];
        den = decay * den + e;
        const float a = sr[o] * (num / (den + WKV_EPS));
        __bf16 hh, ll; split2(a, hh, ll);
        ah[o] = hh; al[o] = ll;
    }
}

// ---------------------------------------------------------------------------
extern "C" void kernel_launch(void* const* d_in, const int* in_sizes, int n_in,
                              void* d_out, int out_size, void* d_ws, size_t ws_size,
                              hipStream_t stream) {
    const float* x     = (const float*)d_in[0];
    const float* ln1_g = (const float*)d_in[1];
    const float* ln1_b = (const float*)d_in[2];
    const float* ln2_g = (const float*)d_in[3];
    const float* ln2_b = (const float*)d_in[4];
    const float* Wr    = (const float*)d_in[5];
    const float* Ww    = (const float*)d_in[6];
    const float* Wk    = (const float*)d_in[7];
    const float* Wv    = (const float*)d_in[8];
    const float* Wo    = (const float*)d_in[9];
    const float* tdcy  = (const float*)d_in[10];
    // d_in[11] = time_first (unused by reference)
    const float* Wkey  = (const float*)d_in[12];
    const float* Wval  = (const float*)d_in[13];
    const float* Wrec  = (const float*)d_in[14];

    const size_t A  = (size_t)MROWS * DD;     // 8,388,608
    const size_t AF = (size_t)MROWS * FF;     // 33,554,432
    const size_t WD = (size_t)DD * DD;
    const size_t WF = (size_t)DD * FF;
    const size_t CH = (size_t)BB * SC * DD;   // 131,072 chunk states

    char* ws = (char*)d_ws;
    size_t off = 0;
    auto alloc = [&](size_t bytes) -> char* {
        off = (off + 255) & ~(size_t)255;
        char* p = ws + off;
        off += bytes;
        return p;
    };
    // bf16 weight splits
    __bf16 *Wrh = (__bf16*)alloc(WD*2),  *Wrl = (__bf16*)alloc(WD*2);
    __bf16 *Wwh = (__bf16*)alloc(WD*2),  *Wwl = (__bf16*)alloc(WD*2);
    __bf16 *Wkh = (__bf16*)alloc(WD*2),  *Wkl = (__bf16*)alloc(WD*2);
    __bf16 *Wvh = (__bf16*)alloc(WD*2),  *Wvl = (__bf16*)alloc(WD*2);
    __bf16 *Woh = (__bf16*)alloc(WD*2),  *Wol = (__bf16*)alloc(WD*2);
    __bf16 *Wch = (__bf16*)alloc(WD*2),  *Wcl = (__bf16*)alloc(WD*2);   // Wrec
    __bf16 *Wyh = (__bf16*)alloc(WF*2),  *Wyl = (__bf16*)alloc(WF*2);   // Wkey
    __bf16 *Wzh = (__bf16*)alloc(WF*2),  *Wzl = (__bf16*)alloc(WF*2);   // Wval
    // activation splits (reused: xn1 -> a -> xn2)
    __bf16 *xh  = (__bf16*)alloc(A*2),   *xl  = (__bf16*)alloc(A*2);
    __bf16 *k2h = (__bf16*)alloc(AF*2),  *k2l = (__bf16*)alloc(AF*2);
    // fp32 scratch (aliased: kbuf->x1, sr->r2)
    float *sr   = (float*)alloc(A*4);
    float *kbuf = (float*)alloc(A*4);
    float *ewk  = (float*)alloc(A*4);
    float *vbuf = (float*)alloc(A*4);
    // chunked-scan scratch
    float *locN  = (float*)alloc(CH*4), *locD  = (float*)alloc(CH*4);
    float *initN = (float*)alloc(CH*4), *initD = (float*)alloc(CH*4);
    float *x1 = kbuf;
    float *r2 = sr;

    float* out_x     = (float*)d_out;
    float* out_state = out_x + A;

    auto split = [&](const float* s, __bf16* h, __bf16* l, size_t n) {
        k_split<<<dim3((unsigned)((n + 255) / 256)), 256, 0, stream>>>(s, h, l, (int)n);
    };
    // 1) weight prep
    split(Wr,  Wrh, Wrl, WD);  split(Ww,  Wwh, Wwl, WD);
    split(Wk,  Wkh, Wkl, WD);  split(Wv,  Wvh, Wvl, WD);
    split(Wo,  Woh, Wol, WD);  split(Wrec,Wch, Wcl, WD);
    split(Wkey,Wyh, Wyl, WF);  split(Wval,Wzh, Wzl, WF);

    // 2) LN1 + split
    k_ln_split<<<MROWS, 256, 0, stream>>>(x, ln1_g, ln1_b, xh, xl);

    const dim3 gD(DD / BN, MROWS / BM);   // (8, 64)
    const dim3 gF(FF / BN, MROWS / BM);   // (32, 64)

    // 3) time-mix projections
    k_gemm<EP_SIGMOID><<<gD, 256, 0, stream>>>(xh, xl, Wrh, Wrl, DD, DD,
                                               sr,  nullptr, nullptr, nullptr, nullptr);
    k_gemm<EP_STORE  ><<<gD, 256, 0, stream>>>(xh, xl, Wkh, Wkl, DD, DD,
                                               kbuf, nullptr, nullptr, nullptr, nullptr);
    k_gemm<EP_EWK    ><<<gD, 256, 0, stream>>>(xh, xl, Wwh, Wwl, DD, DD,
                                               ewk, nullptr, nullptr, kbuf, nullptr);
    k_gemm<EP_STORE  ><<<gD, 256, 0, stream>>>(xh, xl, Wvh, Wvl, DD, DD,
                                               vbuf, nullptr, nullptr, nullptr, nullptr);

    // 4) WKV chunked scan (final state -> d_out tail; a -> xh/xl)
    k_scan_phase1<<<(BB*SC*DD)/256, 256, 0, stream>>>(ewk, vbuf, tdcy, locN, locD);
    k_scan_phase2<<<(BB*DD)/256,    256, 0, stream>>>(locN, locD, tdcy, initN, initD, out_state);
    k_scan_phase3<<<(BB*SC*DD)/256, 256, 0, stream>>>(ewk, vbuf, sr, tdcy, initN, initD, xh, xl);

    // 5) output projection + residual: x1 = x + (a @ Wo^T)
    k_gemm<EP_ADDX><<<gD, 256, 0, stream>>>(xh, xl, Woh, Wol, DD, DD,
                                            x1, nullptr, nullptr, x, nullptr);

    // 6) LN2 + split
    k_ln_split<<<MROWS, 256, 0, stream>>>(x1, ln2_g, ln2_b, xh, xl);

    // 7) channel mix
    k_gemm<EP_RELU2SPLIT><<<gF, 256, 0, stream>>>(xh, xl, Wyh, Wyl, FF, DD,
                                                  nullptr, k2h, k2l, nullptr, nullptr);
    k_gemm<EP_SIGMOID   ><<<gD, 256, 0, stream>>>(xh, xl, Wch, Wcl, DD, DD,
                                                  r2, nullptr, nullptr, nullptr, nullptr);
    k_gemm<EP_FINAL     ><<<gD, 256, 0, stream>>>(k2h, k2l, Wzh, Wzl, DD, FF,
                                                  out_x, nullptr, nullptr, x1, r2);
    (void)in_sizes; (void)n_in; (void)out_size; (void)ws_size;
}
